// Mamba2_89532888252561
// MI455X (gfx1250) — matrix operand order repaired and checked by reading, not compiled, not run on hardware
//
#include <hip/hip_runtime.h>
#include <hip/hip_bf16.h>
#include <cmath>

// ---------------- Mamba2 forward on gfx1250 (CDNA5, wave32, WMMA) -------------
// Big GEMMs: bf16x3 split-precision on V_WMMA_F32_16X16X32_BF16 (fp32-level
// accuracy at ~2.7x the fp32-WMMA rate). SSM chunk math: exact fp32 on
// V_WMMA_F32_16X16X4_F32.

#define MROWS     4096      // B*L
#define DMODEL    2048
#define DINNER    4096
#define NHEADS    64
#define HEADDIM   64
#define DSTATE    128
#define DCONV     4
#define CHUNKLEN  256
#define NCHUNK    16        // B * (L/CHUNK) total chunks
#define DINPROJ   8512
#define CONVDIM   4352
#define SEQL      2048

typedef __attribute__((ext_vector_type(2)))  float  v2f;
typedef __attribute__((ext_vector_type(8)))  float  v8f;
typedef __bf16 bf16;
typedef __attribute__((ext_vector_type(8)))  __bf16 v8bf;
typedef __attribute__((ext_vector_type(16))) __bf16 v16bf;

__device__ __forceinline__ v8f wmma_f32(v2f a, v2f b, v8f c) {
  return __builtin_amdgcn_wmma_f32_16x16x4_f32(false, a, false, b, (short)0, c,
                                               false, false);
}
__device__ __forceinline__ v8f wmma_bf16(v16bf a, v16bf b, v8f c) {
  return __builtin_amdgcn_wmma_f32_16x16x32_bf16(false, a, false, b, (short)0, c,
                                                 false, false);
}
__device__ __forceinline__ v16bf cat16(v8bf l, v8bf h) {
  return __builtin_shufflevector(l, h, 0, 1, 2, 3, 4, 5, 6, 7, 8, 9, 10, 11, 12,
                                 13, 14, 15);
}
__device__ __forceinline__ float sigmoidf_(float x) { return 1.f / (1.f + __expf(-x)); }

// ---------------------------------------------------------------------------
// fp32 -> (hi, lo) bf16 split:  x ~= hi + lo,  lo captures next 8 mantissa bits
// ---------------------------------------------------------------------------
__global__ void split_bf16_kernel(const float* __restrict__ src,
                                  bf16* __restrict__ hi, bf16* __restrict__ lo,
                                  int n) {
  const int i = blockIdx.x * 256 + threadIdx.x;
  if (i < n) {
    const float x = src[i];
    const bf16 h = (bf16)x;
    hi[i] = h;
    lo[i] = (bf16)(x - (float)h);
  }
}

// ---------------------------------------------------------------------------
// bf16x3 GEMM: C[M,N] = A[M,K] @ W[N,K]^T with A,W pre-split into hi/lo bf16.
// One wave per 64x32 tile; 16x16x32 bf16 WMMA, 3 products per tile-pair.
// K multiple of 32; M mult of 64; N mult of 32.
// ---------------------------------------------------------------------------
__global__ void __launch_bounds__(32)
gemm_nt_bf16x3(const bf16* __restrict__ Ahi, const bf16* __restrict__ Alo,
               const bf16* __restrict__ Whi, const bf16* __restrict__ Wlo,
               float* __restrict__ C, int M, int N, int K) {
  const int lane = threadIdx.x & 31;
  const int half = lane >> 4;
  const int r    = lane & 15;
  const int m0 = blockIdx.y * 64;
  const int n0 = blockIdx.x * 32;

  v8f acc[4][2] = {};
  for (int k = 0; k < K; k += 32) {
    v16bf ah[4], al[4], bh[2], bl[2];
#pragma unroll
    for (int mi = 0; mi < 4; ++mi) {   // A rows m0+16*mi+r ; K runs 8h & 16+8h
      const size_t ro = (size_t)(m0 + 16 * mi + r) * K + k;
      ah[mi] = cat16(*(const v8bf*)(Ahi + ro + 8 * half),
                     *(const v8bf*)(Ahi + ro + 16 + 8 * half));
      al[mi] = cat16(*(const v8bf*)(Alo + ro + 8 * half),
                     *(const v8bf*)(Alo + ro + 16 + 8 * half));
    }
#pragma unroll
    for (int ni = 0; ni < 2; ++ni) {   // B col n0+16*ni+r ; K runs 8h & 16+8h
      const size_t ro = (size_t)(n0 + 16 * ni + r) * K + k + 8 * half;
      bh[ni] = cat16(*(const v8bf*)(Whi + ro), *(const v8bf*)(Whi + ro + 16));
      bl[ni] = cat16(*(const v8bf*)(Wlo + ro), *(const v8bf*)(Wlo + ro + 16));
    }
#pragma unroll
    for (int mi = 0; mi < 4; ++mi)
#pragma unroll
      for (int ni = 0; ni < 2; ++ni) {
        acc[mi][ni] = wmma_bf16(ah[mi], bh[ni], acc[mi][ni]);  // hi*hi
        acc[mi][ni] = wmma_bf16(al[mi], bh[ni], acc[mi][ni]);  // lo*hi
        acc[mi][ni] = wmma_bf16(ah[mi], bl[ni], acc[mi][ni]);  // hi*lo
      }
  }
#pragma unroll
  for (int mi = 0; mi < 4; ++mi)
#pragma unroll
    for (int ni = 0; ni < 2; ++ni)
#pragma unroll
      for (int i = 0; i < 8; ++i)
        C[(size_t)(m0 + 16 * mi + i + 8 * half) * N + (n0 + 16 * ni + r)] =
            acc[mi][ni][i];
}

// ---------------------------------------------------------------------------
// Depthwise causal conv (D_CONV=4) + SiLU over the xBC slice of zxbcdt.
// ---------------------------------------------------------------------------
__global__ void conv_silu_kernel(const float* __restrict__ zxbcdt,
                                 const float* __restrict__ conv_w,
                                 const float* __restrict__ conv_b,
                                 float* __restrict__ xBC_act) {
  const int c = blockIdx.x * 256 + threadIdx.x;   // 0..CONVDIM-1
  const int m = blockIdx.y;                       // 0..MROWS-1
  const int l = m & (SEQL - 1);
  float acc = conv_b[c];
#pragma unroll
  for (int k = 0; k < DCONV; ++k) {
    const int ls = l + k - (DCONV - 1);
    const float xv = (ls >= 0)
        ? zxbcdt[(size_t)(m + k - (DCONV - 1)) * DINPROJ + DINNER + c] : 0.f;
    acc += xv * conv_w[c * DCONV + k];
  }
  xBC_act[(size_t)m * CONVDIM + c] = acc * sigmoidf_(acc);
}

// dt = softplus(dt_raw + dt_bias),  dA = dt * (-exp(A_log))
__global__ void dt_kernel(const float* __restrict__ zxbcdt,
                          const float* __restrict__ dt_bias,
                          const float* __restrict__ A_log,
                          float* __restrict__ dt, float* __restrict__ dA) {
  const int idx = blockIdx.x * 256 + threadIdx.x;   // MROWS*NHEADS
  const int m = idx >> 6, h = idx & 63;
  const float x = zxbcdt[(size_t)m * DINPROJ + (DINNER + CONVDIM) + h] + dt_bias[h];
  const float d = (x > 20.f) ? x : log1pf(__expf(x));
  dt[idx] = d;
  dA[idx] = d * (-__expf(A_log[h]));
}

// xdt[m, d] = x[m, d] * dt[m, h(d)]
__global__ void xdt_kernel(const float* __restrict__ xBC_act,
                           const float* __restrict__ dt,
                           float* __restrict__ xdt) {
  const size_t idx = (size_t)blockIdx.x * 256 + threadIdx.x;  // MROWS*DINNER
  const int m = (int)(idx >> 12), d = (int)(idx & 4095), h = d >> 6;
  xdt[idx] = xBC_act[(size_t)m * CONVDIM + d] * dt[m * NHEADS + h];
}

// Per-chunk cumsum of dA along l (256), per (chunk, head).
__global__ void cumsum_kernel(const float* __restrict__ dA, float* __restrict__ Acs) {
  const int bc = blockIdx.x;          // 0..NCHUNK-1
  const int h  = threadIdx.x;         // 0..63
  const size_t base = (size_t)bc * CHUNKLEN * NHEADS + h;
  float s = 0.f;
  for (int l = 0; l < CHUNKLEN; ++l) {
    s += dA[base + (size_t)l * NHEADS];
    Acs[base + (size_t)l * NHEADS] = s;
  }
}

// ---------------------------------------------------------------------------
// Per-chunk states[p,n] = sum_l xdt[l,p]*exp(Acs_last-Acs[l]) * B[l,n]
// One block (4 waves) per (chunk, head); wave w covers n in [32w, 32w+32).
// ---------------------------------------------------------------------------
__global__ void __launch_bounds__(128)
states_kernel(const float* __restrict__ xBC_act, const float* __restrict__ xdt,
              const float* __restrict__ Acs, float* __restrict__ states) {
  const int blk = blockIdx.x;                 // bc*64 + h
  const int h = blk & 63, bc = blk >> 6;
  const int m0 = bc * CHUNKLEN;
  const int tid = threadIdx.x;
  __shared__ float dec_sh[CHUNKLEN];
  const float last = Acs[(size_t)(m0 + CHUNKLEN - 1) * NHEADS + h];
  for (int i = tid; i < CHUNKLEN; i += 128)
    dec_sh[i] = __expf(last - Acs[(size_t)(m0 + i) * NHEADS + h]);
  __syncthreads();

  const int w = tid >> 5, lane = tid & 31, half = lane >> 4, r = lane & 15;
  const int n0 = w * 32;
  v8f acc[4][2] = {};
  for (int k = 0; k < CHUNKLEN; k += 4) {
    const int ka = k + 2 * half;
    const float d0 = dec_sh[ka], d1 = dec_sh[ka + 1];
    v2f a[4], b[2];
#pragma unroll
    for (int mi = 0; mi < 4; ++mi) {            // p = 16*mi + r
      const float* xp = xdt + (size_t)(m0 + ka) * DINNER + h * HEADDIM + 16 * mi + r;
      a[mi].x = xp[0] * d0; a[mi].y = xp[DINNER] * d1;
    }
#pragma unroll
    for (int ni = 0; ni < 2; ++ni) {            // n = n0 + 16*ni + r
      const float* bp = xBC_act + (size_t)(m0 + ka) * CONVDIM + DINNER + n0 + 16 * ni + r;
      b[ni].x = bp[0]; b[ni].y = bp[CONVDIM];
    }
#pragma unroll
    for (int mi = 0; mi < 4; ++mi)
#pragma unroll
      for (int ni = 0; ni < 2; ++ni)
        acc[mi][ni] = wmma_f32(a[mi], b[ni], acc[mi][ni]);
  }
  float* S = states + (size_t)blk * HEADDIM * DSTATE;
#pragma unroll
  for (int mi = 0; mi < 4; ++mi)
#pragma unroll
    for (int ni = 0; ni < 2; ++ni)
#pragma unroll
      for (int i = 0; i < 8; ++i)
        S[(size_t)(16 * mi + i + 8 * half) * DSTATE + n0 + 16 * ni + r] = acc[mi][ni][i];
}

// Sequential inter-chunk scan: prev[c] = h before chunk c; h = decay*h + states.
__global__ void scan_kernel(const float* __restrict__ states,
                            const float* __restrict__ Acs,
                            float* __restrict__ prev) {
  const size_t idx = (size_t)blockIdx.x * 256 + threadIdx.x; // B * 64*64*128
  const int b = (int)(idx / (NHEADS * HEADDIM * DSTATE));
  const size_t rem = idx % (NHEADS * HEADDIM * DSTATE);      // h*8192 + p*128 + n
  const int h = (int)(rem / (HEADDIM * DSTATE));
  float hstate = 0.f;
  for (int c = 0; c < SEQL / CHUNKLEN; ++c) {
    const int bc = b * (SEQL / CHUNKLEN) + c;
    const size_t off = (size_t)bc * NHEADS * HEADDIM * DSTATE + rem;
    prev[off] = hstate;
    const float cd = __expf(Acs[(size_t)(bc * CHUNKLEN + CHUNKLEN - 1) * NHEADS + h]);
    hstate = cd * hstate + states[off];
  }
}

// ---------------------------------------------------------------------------
// Per-chunk, per-head:  y = (mask(exp(dAcs)) ∘ (C Bᵀ)) Xdt + exp(Acs)·(C Pᵀ) + D·x
// Block = 128 threads (4 waves); wave w handles l-tiles {w, w+4, w+8, w+12}.
// Score tile S goes D-layout -> LDS -> A-layout (per-wave staging, in-order DS).
// ---------------------------------------------------------------------------
__global__ void __launch_bounds__(128)
ssm_chunk_kernel(const float* __restrict__ xBC_act, const float* __restrict__ xdt,
                 const float* __restrict__ Acs, const float* __restrict__ prev,
                 const float* __restrict__ D_param, float* __restrict__ y) {
  const int blk = blockIdx.x;             // bc*64 + h
  const int h = blk & 63, bc = blk >> 6;
  const int m0 = bc * CHUNKLEN;
  const int tid = threadIdx.x;
  __shared__ float acs_sh[CHUNKLEN];
  __shared__ float s_stage[4][16 * 16];
  for (int i = tid; i < CHUNKLEN; i += 128)
    acs_sh[i] = Acs[(size_t)(m0 + i) * NHEADS + h];
  __syncthreads();

  const int w = tid >> 5, lane = tid & 31, half = lane >> 4, r = lane & 15;
  const float* Ph = prev + (size_t)blk * HEADDIM * DSTATE;   // [p][n]
  const float dpar = D_param[h];

  for (int li = w; li < CHUNKLEN / 16; li += 4) {
    const int l0 = li * 16;
    v8f acc[4] = {};        // Y_diag rows l0..l0+15, cols p = 0..63
    for (int si = 0; si <= li; ++si) {
      const int s0 = si * 16;
      // ---- G = C[l-tile] (16x128) @ B[s-tile]^T, K = D_STATE ----
      v8f g = {};
      for (int k = 0; k < DSTATE; k += 4) {
        const int ka = k + 2 * half;
        v2f a, b;
        const float* cp = xBC_act + (size_t)(m0 + l0 + r) * CONVDIM + DINNER + DSTATE + ka;
        a.x = cp[0]; a.y = cp[1];
        const float* bp = xBC_act + (size_t)(m0 + s0 + r) * CONVDIM + DINNER + ka;
        b.x = bp[0]; b.y = bp[1];
        g = wmma_f32(a, b, g);
      }
      // ---- mask + segment-exp, stage to LDS (row-major S tile) ----
#pragma unroll
      for (int i = 0; i < 8; ++i) {
        const int lrow = l0 + i + 8 * half;
        const int scol = s0 + r;
        float v = 0.f;
        if (scol <= lrow) v = g[i] * __expf(acs_sh[lrow] - acs_sh[scol]);
        s_stage[w][(i + 8 * half) * 16 + r] = v;
      }
      // ---- Y += S (16x16) @ Xdt[s-tile] (16x64) ----
#pragma unroll
      for (int kk = 0; kk < 16; kk += 4) {
        v2f a;
        a.x = s_stage[w][r * 16 + kk + 2 * half];
        a.y = s_stage[w][r * 16 + kk + 2 * half + 1];
#pragma unroll
        for (int ni = 0; ni < 4; ++ni) {
          v2f b;
          const float* xp = xdt + (size_t)(m0 + s0 + kk + 2 * half) * DINNER
                              + h * HEADDIM + 16 * ni + r;
          b.x = xp[0]; b.y = xp[DINNER];
          acc[ni] = wmma_f32(a, b, acc[ni]);
        }
      }
    }
    // ---- Y_off tile: C[l-tile] (16x128) @ prev^T (128x64), then row-scale ----
    v8f acc2[4] = {};
    for (int k = 0; k < DSTATE; k += 4) {
      const int ka = k + 2 * half;
      v2f a;
      const float* cp = xBC_act + (size_t)(m0 + l0 + r) * CONVDIM + DINNER + DSTATE + ka;
      a.x = cp[0]; a.y = cp[1];
#pragma unroll
      for (int ni = 0; ni < 4; ++ni) {
        v2f b;
        const float* pp = Ph + (size_t)(16 * ni + r) * DSTATE + ka;
        b.x = pp[0]; b.y = pp[1];
        acc2[ni] = wmma_f32(a, b, acc2[ni]);
      }
    }
    // ---- combine + D*x, store ----
#pragma unroll
    for (int i = 0; i < 8; ++i) {
      const int lrow = l0 + i + 8 * half;
      const float e = __expf(acs_sh[lrow]);
      const int m = m0 + lrow;
#pragma unroll
      for (int ni = 0; ni < 4; ++ni) {
        const int p = 16 * ni + r;
        const int d = h * HEADDIM + p;
        const float xv = xBC_act[(size_t)m * CONVDIM + d];
        y[(size_t)m * DINNER + d] = acc[ni][i] + e * acc2[ni][i] + dpar * xv;
      }
    }
  }
}

// Gated RMSNorm: yg = y*silu(z); out = yg * rsqrt(mean(yg^2)+eps) * rms_w
__global__ void __launch_bounds__(256)
gatenorm_kernel(const float* __restrict__ y, const float* __restrict__ zxbcdt,
                const float* __restrict__ rms_w, float* __restrict__ out) {
  const int m = blockIdx.x, tid = threadIdx.x;
  __shared__ float yg_sh[DINNER];
  __shared__ float rbuf[256];
  float ss = 0.f;
  for (int i = tid; i < DINNER; i += 256) {
    const float z = zxbcdt[(size_t)m * DINPROJ + i];
    const float v = y[(size_t)m * DINNER + i] * (z * sigmoidf_(z));
    yg_sh[i] = v;
    ss += v * v;
  }
  rbuf[tid] = ss; __syncthreads();
  for (int s = 128; s > 0; s >>= 1) {
    if (tid < s) rbuf[tid] += rbuf[tid + s];
    __syncthreads();
  }
  const float scale = rsqrtf(rbuf[0] / (float)DINNER + 1e-5f);
  for (int i = tid; i < DINNER; i += 256)
    out[(size_t)m * DINNER + i] = yg_sh[i] * scale * rms_w[i];
}

// ---------------------------------------------------------------------------
extern "C" void kernel_launch(void* const* d_in, const int* in_sizes, int n_in,
                              void* d_out, int out_size, void* d_ws, size_t ws_size,
                              hipStream_t stream) {
  const float* u       = (const float*)d_in[0];
  const float* W_in    = (const float*)d_in[1];
  const float* conv_w  = (const float*)d_in[2];
  const float* conv_b  = (const float*)d_in[3];
  const float* dt_bias = (const float*)d_in[4];
  const float* A_log   = (const float*)d_in[5];
  const float* D_param = (const float*)d_in[6];
  const float* rms_w   = (const float*)d_in[7];
  const float* W_out   = (const float*)d_in[8];
  float* out = (float*)d_out;

  float* ws = (float*)d_ws;
  // fp32 workspace layout
  float* zxbcdt  = ws;                                           // 4096*8512
  float* xBC_act = zxbcdt  + (size_t)MROWS * DINPROJ;            // 4096*4352
  float* dt      = xBC_act + (size_t)MROWS * CONVDIM;            // 4096*64
  float* dA      = dt      + (size_t)MROWS * NHEADS;             // 4096*64
  float* Acs     = dA      + (size_t)MROWS * NHEADS;             // 4096*64
  float* xdt     = Acs     + (size_t)MROWS * NHEADS;             // 4096*4096
  float* states  = xdt     + (size_t)MROWS * DINNER;             // 16*64*64*128
  float* prev    = states  + (size_t)NCHUNK * NHEADS * HEADDIM * DSTATE;
  float* ybuf    = prev    + (size_t)NCHUNK * NHEADS * HEADDIM * DSTATE; // 4096*4096
  float* ynorm   = xdt;    // reuse: xdt dead after ssm_chunk_kernel
  // bf16 split scratch (reused between in_proj and out_proj phases)
  bf16* bfscr = (bf16*)(ybuf + (size_t)MROWS * DINNER);
  const int NU = MROWS * DMODEL;            // 8,388,608
  const int NWIN = DINPROJ * DMODEL;        // 17,432,576
  const int NY = MROWS * DINNER;            // 16,777,216
  const int NWOUT = DMODEL * DINNER;        // 8,388,608
  bf16* u_hi = bfscr;              bf16* u_lo = u_hi + NU;
  bf16* wi_hi = u_lo + NU;         bf16* wi_lo = wi_hi + NWIN;
  bf16* y_hi = bfscr;              bf16* y_lo = y_hi + NY;       // phase-2 reuse
  bf16* wo_hi = y_lo + NY;         bf16* wo_lo = wo_hi + NWOUT;

  // 1) split + in_proj GEMM (bf16x3): zxbcdt = u @ W_in^T
  split_bf16_kernel<<<(NU + 255) / 256, 256, 0, stream>>>(u, u_hi, u_lo, NU);
  split_bf16_kernel<<<(NWIN + 255) / 256, 256, 0, stream>>>(W_in, wi_hi, wi_lo, NWIN);
  gemm_nt_bf16x3<<<dim3(DINPROJ / 32, MROWS / 64), 32, 0, stream>>>(
      u_hi, u_lo, wi_hi, wi_lo, zxbcdt, MROWS, DINPROJ, DMODEL);

  // 2) causal depthwise conv + SiLU
  conv_silu_kernel<<<dim3(CONVDIM / 256, MROWS), 256, 0, stream>>>(
      zxbcdt, conv_w, conv_b, xBC_act);

  // 3) dt / dA
  dt_kernel<<<(MROWS * NHEADS) / 256, 256, 0, stream>>>(zxbcdt, dt_bias, A_log, dt, dA);

  // 4) xdt = x * dt
  xdt_kernel<<<(int)(((size_t)MROWS * DINNER) / 256), 256, 0, stream>>>(xBC_act, dt, xdt);

  // 5) per-chunk cumsum of dA
  cumsum_kernel<<<NCHUNK, NHEADS, 0, stream>>>(dA, Acs);

  // 6) chunk states (fp32 WMMA)
  states_kernel<<<NCHUNK * NHEADS, 128, 0, stream>>>(xBC_act, xdt, Acs, states);

  // 7) sequential inter-chunk scan
  scan_kernel<<<(int)(((size_t)2 * NHEADS * HEADDIM * DSTATE) / 256), 256, 0, stream>>>(
      states, Acs, prev);

  // 8) chunk-local attention-like SSM output (fp32 WMMA) + Y_off + D*x
  ssm_chunk_kernel<<<NCHUNK * NHEADS, 128, 0, stream>>>(
      xBC_act, xdt, Acs, prev, D_param, ybuf);

  // 9) gated RMSNorm
  gatenorm_kernel<<<MROWS, 256, 0, stream>>>(ybuf, zxbcdt, rms_w, ynorm);

  // 10) split + out_proj GEMM (bf16x3): out = ynorm @ W_out^T
  split_bf16_kernel<<<(NY + 255) / 256, 256, 0, stream>>>(ynorm, y_hi, y_lo, NY);
  split_bf16_kernel<<<(NWOUT + 255) / 256, 256, 0, stream>>>(W_out, wo_hi, wo_lo, NWOUT);
  gemm_nt_bf16x3<<<dim3(DMODEL / 32, MROWS / 64), 32, 0, stream>>>(
      y_hi, y_lo, wo_hi, wo_lo, out, MROWS, DMODEL, DINNER);
}
